// VQVAE_44169443672878
// MI455X (gfx1250) — matrix operand order repaired and checked by reading, not compiled, not run on hardware
//
#include <hip/hip_runtime.h>
#include <hip/hip_bf16.h>
#include <math.h>

#define KNUM 8192
#define DDIM 512
#define NVEC 16384   // B*M = 64*256
#define BETA 0.25f
#define MTILE 4      // row tiles (16 rows each) per block sharing one B stream

typedef __attribute__((ext_vector_type(16))) __bf16 v16bf;
typedef __attribute__((ext_vector_type(8)))  __bf16 v8bf;
typedef __attribute__((ext_vector_type(8)))  float  v8f;

static __device__ __forceinline__ unsigned short f32_to_bf16(float f) {
  unsigned int u = __float_as_uint(f);
  u += 0x7FFFu + ((u >> 16) & 1u);   // round-to-nearest-even
  return (unsigned short)(u >> 16);
}

// ---------------------------------------------------------------------------
// Kernel 1: zero the usage histogram and loss accumulator
// ---------------------------------------------------------------------------
__global__ void vq_init(float* __restrict__ counts, float* __restrict__ loss_acc) {
  int i = blockIdx.x * blockDim.x + threadIdx.x;
  if (i < KNUM) counts[i] = 0.0f;
  if (i == 0) *loss_acc = 0.0f;
}

// ---------------------------------------------------------------------------
// Kernel 2: codebook -> bf16 (RNE) + fp32 row norms ||e||^2
// ---------------------------------------------------------------------------
__global__ void vq_prep_codebook(const float* __restrict__ emb,
                                 unsigned short* __restrict__ ebf,
                                 float* __restrict__ enorm) {
  __shared__ float red[256];
  const int row = blockIdx.x;
  const float* src = emb + (size_t)row * DDIM;
  unsigned short* dst = ebf + (size_t)row * DDIM;
  float s = 0.0f;
  for (int d = threadIdx.x; d < DDIM; d += blockDim.x) {
    float f = src[d];
    dst[d] = f32_to_bf16(f);
    s += f * f;
  }
  red[threadIdx.x] = s;
  __syncthreads();
  for (int off = 128; off > 0; off >>= 1) {
    if ((int)threadIdx.x < off) red[threadIdx.x] += red[threadIdx.x + off];
    __syncthreads();
  }
  if (threadIdx.x == 0) enorm[row] = red[0];
}

// ---------------------------------------------------------------------------
// Kernel 3: WMMA-based nearest-code search.
// Block = 128 threads (4 waves), 64 rows (MTILE=4 row tiles of 16).
// A fragments: reloaded from the LDS bf16 strip for every WMMA. B fragments: global,
// 32 B/lane/WMMA, double-buffered across ct iterations so each 16 KB B batch
// prefetches under the previous batch's 64-WMMA shadow.
// distance ordering uses d(n) = ||e_n||^2 - 2 * <x, e_n>   (||x||^2 dropped)
// ---------------------------------------------------------------------------
__global__ __launch_bounds__(128) void vq_argmin(
    const float* __restrict__ z_e,
    const unsigned short* __restrict__ ebf,
    const float* __restrict__ enorm,
    int*   __restrict__ idx_out,
    float* __restrict__ idx_f_out,
    float* __restrict__ counts) {
  __shared__ __align__(16) unsigned short Abf[16 * MTILE][DDIM + 8];
  __shared__ float wv[4][16 * MTILE];
  __shared__ int   wi[4][16 * MTILE];

  const int row0 = blockIdx.x * (16 * MTILE);
  const float* X = z_e + (size_t)row0 * DDIM;
  // stage 64x512 strip of z_e into LDS as bf16
  for (int i = threadIdx.x; i < 16 * MTILE * DDIM; i += blockDim.x) {
    int r = i >> 9, d = i & (DDIM - 1);
    Abf[r][d] = f32_to_bf16(X[(size_t)r * DDIM + d]);
  }
  __syncthreads();

  const int wave = threadIdx.x >> 5;
  const int lane = threadIdx.x & 31;
  const int half = lane >> 4;      // lane half selects K sub-range (contiguous 16-K halves, same for A and B)
  const int l16  = lane & 15;      // A row / B column within tile
  const int akoff = half * 16;     // A: lanes 0-15 get K 0-15, 16-31 get 16-31

  float bestv[MTILE][8];
  int   besti[MTILE][8];
#pragma unroll
  for (int t = 0; t < MTILE; ++t)
#pragma unroll
    for (int v = 0; v < 8; ++v) { bestv[t][v] = 3.4e38f; besti[t][v] = 0x7fffffff; }

  // per-column-tile compute: 16 k-steps x MTILE WMMAs sharing B, then argmin
  auto compute_tile = [&](const v16bf* bbuf, int col) {
    v8f acc[MTILE];
#pragma unroll
    for (int t = 0; t < MTILE; ++t) acc[t] = (v8f){};
#pragma unroll
    for (int s = 0; s < 16; ++s) {           // K-depth: 16 x 32 = 512
#pragma unroll
      for (int t = 0; t < MTILE; ++t) {
        union { v16bf v; v8bf h[2]; } a;
        a.h[0] = *(const v8bf*)&Abf[t * 16 + l16][s * 32 + akoff];
        a.h[1] = *(const v8bf*)&Abf[t * 16 + l16][s * 32 + akoff + 8];
        acc[t] = __builtin_amdgcn_wmma_f32_16x16x32_bf16(
            false, a.v, false, bbuf[s], (short)0, acc[t], false, false);
      }
    }
    const float en = enorm[col];
#pragma unroll
    for (int t = 0; t < MTILE; ++t)
#pragma unroll
      for (int v = 0; v < 8; ++v) {          // D layout: M = v + 8*half, N = l16
        float d = en - 2.0f * acc[t][v];
        if (d < bestv[t][v]) { bestv[t][v] = d; besti[t][v] = col; }
      }
  };

  // wave sweeps column tiles ct = wave, wave+4, ... (128 tiles), unrolled x2
  // with ping-pong B buffers for cross-iteration prefetch.
  const size_t BSTRIDE4 = (size_t)4 * 16 * DDIM;  // elements between ct and ct+4
  const unsigned short* bp = ebf + (size_t)(wave * 16 + l16) * DDIM + half * 16;
  v16bf buf0[16], buf1[16];
#pragma unroll
  for (int s = 0; s < 16; ++s) buf0[s] = *(const v16bf*)(bp + s * 32);

  for (int ct = wave; ct < KNUM / 16; ct += 8) {
    // prefetch ct+4 while computing ct
#pragma unroll
    for (int s = 0; s < 16; ++s) buf1[s] = *(const v16bf*)(bp + BSTRIDE4 + s * 32);
    compute_tile(buf0, ct * 16 + l16);
    // prefetch ct+8 while computing ct+4 (final overshoot lands in adjacent
    // workspace arrays; loaded but never consumed)
#pragma unroll
    for (int s = 0; s < 16; ++s) buf0[s] = *(const v16bf*)(bp + 2 * BSTRIDE4 + s * 32);
    compute_tile(buf1, (ct + 4) * 16 + l16);
    bp += 2 * BSTRIDE4;
  }

  // reduce over the 16 lanes of each half (xor 8/4/2/1 stays within a half)
#pragma unroll
  for (int t = 0; t < MTILE; ++t)
#pragma unroll
    for (int v = 0; v < 8; ++v) {
      float bv = bestv[t][v];
      int   bi = besti[t][v];
      for (int m = 8; m >= 1; m >>= 1) {
        float ov = __shfl_xor(bv, m, 32);
        int   oi = __shfl_xor(bi, m, 32);
        if (ov < bv || (ov == bv && oi < bi)) { bv = ov; bi = oi; }
      }
      if (l16 == 0) {
        wv[wave][t * 16 + v + 8 * half] = bv;
        wi[wave][t * 16 + v + 8 * half] = bi;
      }
    }
  __syncthreads();

  // combine the 4 waves' disjoint column ranges
  if (threadIdx.x < 16 * MTILE) {
    float bv = wv[0][threadIdx.x];
    int   bi = wi[0][threadIdx.x];
    for (int w = 1; w < 4; ++w) {
      float ov = wv[w][threadIdx.x];
      int   oi = wi[w][threadIdx.x];
      if (ov < bv || (ov == bv && oi < bi)) { bv = ov; bi = oi; }
    }
    const int row = row0 + threadIdx.x;
    idx_out[row]   = bi;
    idx_f_out[row] = (float)bi;
    atomicAdd(&counts[bi], 1.0f);
  }
}

// ---------------------------------------------------------------------------
// Kernel 4: gather z_q (== straight-through output value) + commitment loss
// ---------------------------------------------------------------------------
__global__ void vq_gather_loss(const float* __restrict__ z_e,
                               const float* __restrict__ emb,
                               const int* __restrict__ idx,
                               float* __restrict__ zq_out,
                               float* __restrict__ loss_acc) {
  __shared__ float red[256];
  const int row = blockIdx.x;
  const int k = idx[row];
  const float* e = emb + (size_t)k * DDIM;
  const float* x = z_e + (size_t)row * DDIM;
  float* o = zq_out + (size_t)row * DDIM;
  float s = 0.0f;
  for (int d = threadIdx.x; d < DDIM; d += blockDim.x) {
    float q = e[d];
    o[d] = q;                 // z_e + (z_q - z_e) == z_q value-wise
    float df = q - x[d];
    s += df * df;
  }
  red[threadIdx.x] = s;
  __syncthreads();
  for (int off = 128; off > 0; off >>= 1) {
    if ((int)threadIdx.x < off) red[threadIdx.x] += red[threadIdx.x + off];
    __syncthreads();
  }
  if (threadIdx.x == 0) atomicAdd(loss_acc, red[0]);
}

// ---------------------------------------------------------------------------
// Kernel 5: finalize loss scale + perplexity
// ---------------------------------------------------------------------------
__global__ void vq_finalize(const float* __restrict__ counts,
                            const float* __restrict__ loss_acc,
                            float* __restrict__ out_loss,
                            float* __restrict__ out_perp) {
  __shared__ float red[256];
  float h = 0.0f;
  for (int k = threadIdx.x; k < KNUM; k += blockDim.x) {
    float p = counts[k] * (1.0f / (float)NVEC);
    p = fmaxf(p, 1e-12f);
    h += p * logf(p);
  }
  red[threadIdx.x] = h;
  __syncthreads();
  for (int off = 128; off > 0; off >>= 1) {
    if ((int)threadIdx.x < off) red[threadIdx.x] += red[threadIdx.x + off];
    __syncthreads();
  }
  if (threadIdx.x == 0) {
    *out_loss = BETA * loss_acc[0] / (float)((size_t)NVEC * DDIM);
    *out_perp = expf(-red[0]);
  }
}

// ---------------------------------------------------------------------------
extern "C" void kernel_launch(void* const* d_in, const int* in_sizes, int n_in,
                              void* d_out, int out_size, void* d_ws, size_t ws_size,
                              hipStream_t stream) {
  const float* z_e = (const float*)d_in[0];   // [64,256,512] f32
  const float* emb = (const float*)d_in[1];   // [8192,512]   f32

  float* out = (float*)d_out;
  float* zq_out   = out;                           // N*D floats
  float* out_loss = out + (size_t)NVEC * DDIM;     // 1
  float* out_idx  = out_loss + 1;                  // N floats
  float* out_perp = out_idx + NVEC;                // 1

  char* ws = (char*)d_ws;
  unsigned short* ebf = (unsigned short*)ws;  ws += (size_t)KNUM * DDIM * sizeof(unsigned short);
  float* enorm   = (float*)ws;                ws += (size_t)KNUM * sizeof(float);
  float* counts  = (float*)ws;                ws += (size_t)KNUM * sizeof(float);
  int*   idxbuf  = (int*)ws;                  ws += (size_t)NVEC * sizeof(int);
  float* loss_acc = (float*)ws;

  vq_init<<<(KNUM + 255) / 256, 256, 0, stream>>>(counts, loss_acc);
  vq_prep_codebook<<<KNUM, 256, 0, stream>>>(emb, ebf, enorm);
  vq_argmin<<<NVEC / (16 * MTILE), 128, 0, stream>>>(z_e, ebf, enorm, idxbuf, out_idx, counts);
  vq_gather_loss<<<NVEC, 256, 0, stream>>>(z_e, emb, idxbuf, zq_out, loss_acc);
  vq_finalize<<<1, 256, 0, stream>>>(counts, loss_acc, out_loss, out_perp);
}
